// NeuralODEAutoencoder_46059229282606
// MI455X (gfx1250) — compile-verified, matrix operand order checked by reading, not run on hardware
//
#include <hip/hip_runtime.h>
#include <hip/hip_bf16.h>
#include <math.h>

typedef __bf16 bf16_t;
typedef bf16_t v16bf __attribute__((ext_vector_type(16)));
typedef bf16_t v8bf  __attribute__((ext_vector_type(8)));
typedef float  v8f   __attribute__((ext_vector_type(8)));

#define BATCH 4096
#define TWIN  100
#define DFEAT 64
#define HENC  256
#define LAT   64

// ---------------- device helpers ----------------

__device__ __forceinline__ float sigf(float x){ return 1.0f/(1.0f+__expf(-x)); }
__device__ __forceinline__ float siluf(float x){ return x*sigf(x); }

__device__ __forceinline__ v8f wmma_bf16(v16bf a, v16bf b, v8f c){
  // D = A(16x32 bf16) * B(32x16 bf16) + C(16x16 f32)
  return __builtin_amdgcn_wmma_f32_16x16x32_bf16(false, a, false, b, (short)0, c, false, false);
}

// A fragment: lane L(0..15) holds row m0+L, Ks {k0..k0+15};
// lane L+16 holds row m0+L, Ks {k0+16..k0+31}. One contiguous 32-byte load.
__device__ __forceinline__ v16bf load_a(const bf16_t* A, int m0, int ldk, int k0, int lane){
  return *(const v16bf*)(A + (size_t)(m0 + (lane & 15)) * ldk + k0 + ((lane >> 4) * 16));
}
// B fragments pre-packed: block (kb,nb) = 32 lanes x 16 contiguous bf16
__device__ __forceinline__ v16bf load_bp(const bf16_t* Bp, int kb, int nb, int NB, int lane){
  return *(const v16bf*)(Bp + (((size_t)(kb * NB + nb)) * 32 + lane) * 16);
}

// ---------------- small kernels ----------------

// x (B,T,D) f32 -> (T,B,D) bf16
__global__ void k_cast_x(const float* __restrict__ x, bf16_t* __restrict__ xt){
  size_t i = (size_t)blockIdx.x * 256 + threadIdx.x;
  if (i >= (size_t)BATCH * TWIN * DFEAT) return;
  int d = (int)(i % DFEAT);
  size_t r = i / DFEAT;
  int t = (int)(r % TWIN);
  int b = (int)(r / TWIN);
  xt[((size_t)t * BATCH + b) * DFEAT + d] = (bf16_t)x[i];
}

// W (N rows, K cols) f32 -> packed WMMA-B fragments (Kpad x N) bf16, zero-padded K
__global__ void k_pack_b(const float* __restrict__ W, bf16_t* __restrict__ Bp,
                         int N, int K, int Kpad){
  int NB = N / 16;
  size_t total = (size_t)(Kpad / 32) * NB * 512;
  size_t i = (size_t)blockIdx.x * 256 + threadIdx.x;
  if (i >= total) return;
  int e = (int)(i & 15);
  int lane = (int)((i >> 4) & 31);
  size_t blk = i >> 9;
  int nb = (int)(blk % NB);
  int kb = (int)(blk / NB);
  int n = nb * 16 + (lane & 15);
  int k = kb * 32 + ((lane >> 4) * 16) + e;
  float v = (k < K) ? W[(size_t)n * K + k] : 0.0f;
  Bp[i] = (bf16_t)v;
}

__global__ void k_cast(const float* __restrict__ in, bf16_t* __restrict__ out, size_t n){
  size_t i = (size_t)blockIdx.x * 256 + threadIdx.x;
  if (i < n) out[i] = (bf16_t)in[i];
}

// zt = bf16[ z + alpha*k | t | 0-pad ]  (B x 96)
__global__ void k_make_zt(const float* __restrict__ z, const float* __restrict__ k,
                          float alpha, float tval, bf16_t* __restrict__ zt){
  int i = blockIdx.x * 256 + threadIdx.x;
  if (i >= BATCH * 96) return;
  int c = i % 96, b = i / 96;
  float v;
  if (c < LAT){ v = z[b * LAT + c]; if (k) v += alpha * k[b * LAT + c]; }
  else if (c == LAT){ v = tval; }
  else { v = 0.0f; }
  zt[i] = (bf16_t)v;
}

__global__ void k_rk4(float* __restrict__ z, const float* __restrict__ k1,
                      const float* __restrict__ k2, const float* __restrict__ k3,
                      const float* __restrict__ k4, float dt){
  int i = blockIdx.x * 256 + threadIdx.x;
  if (i >= BATCH * LAT) return;
  z[i] += (dt / 6.0f) * (k1[i] + 2.0f * k2[i] + 2.0f * k3[i] + k4[i]);
}

__global__ void k_vsq(const float* __restrict__ k, float* __restrict__ vsq){
  int b = blockIdx.x * 256 + threadIdx.x;
  if (b >= BATCH) return;
  float s = 0.0f;
#pragma unroll 8
  for (int d = 0; d < LAT; ++d){ float v = k[(size_t)b * LAT + d]; s += v * v; }
  vsq[b] = s;
}

__global__ void k_ke(const float* __restrict__ vsq, float* __restrict__ out, float dt){
  __shared__ float red[256];
  int tid = threadIdx.x;
  float acc = 0.0f;
  for (int b = tid; b < BATCH; b += 256){
    float s = 0.0f;
    for (int i = 0; i < 9; ++i)
      s += 0.5f * (vsq[i * BATCH + b] + vsq[(i + 1) * BATCH + b]) * dt;
    acc += s;
  }
  red[tid] = acc;
  __syncthreads();
  for (int off = 128; off > 0; off >>= 1){
    if (tid < off) red[tid] += red[tid + off];
    __syncthreads();
  }
  if (tid == 0) out[0] = red[0] / (float)BATCH;
}

// ---------------- WMMA GEMM, N % 128 == 0 ----------------
// out = act(A(MxKpad bf16) @ Bpacked(Kpad x N) + bias); M % 64 == 0.
// 256 threads = 8 waves; block tile 64x128; wave tile 32x32. Branch-free WMMA.
__global__ void k_gemm128(const bf16_t* __restrict__ A, const bf16_t* __restrict__ Bp,
                          const float* __restrict__ bias,
                          float* __restrict__ outF, bf16_t* __restrict__ outB,
                          int N, int Kpad, int act){
  const int lane = threadIdx.x & 31;
  const int w = threadIdx.x >> 5;
  const int m0 = blockIdx.y * 64 + (w >> 2) * 32;
  const int n0 = blockIdx.x * 128 + (w & 3) * 32;
  const int NB = N / 16, KB = Kpad / 32;
  const v8f zero = {0.f,0.f,0.f,0.f,0.f,0.f,0.f,0.f};
  v8f c00 = zero, c01 = zero, c10 = zero, c11 = zero;
  for (int kb = 0; kb < KB; ++kb){
    if (kb + 1 < KB)
      __builtin_prefetch(Bp + (((size_t)((kb + 1) * NB + (n0 >> 4))) * 32 + lane) * 16, 0, 3);
    v16bf a0 = load_a(A, m0,      Kpad, kb * 32, lane);
    v16bf a1 = load_a(A, m0 + 16, Kpad, kb * 32, lane);
    v16bf b0 = load_bp(Bp, kb, (n0 >> 4),     NB, lane);
    v16bf b1 = load_bp(Bp, kb, (n0 >> 4) + 1, NB, lane);
    c00 = wmma_bf16(a0, b0, c00);
    c10 = wmma_bf16(a1, b0, c10);
    c01 = wmma_bf16(a0, b1, c01);
    c11 = wmma_bf16(a1, b1, c11);
  }
#pragma unroll
  for (int j = 0; j < 2; ++j){
    int n = n0 + j * 16 + (lane & 15);
    float bv = bias ? bias[n] : 0.0f;
#pragma unroll
    for (int i = 0; i < 2; ++i){
      v8f cc = (i == 0) ? (j == 0 ? c00 : c01) : (j == 0 ? c10 : c11);
#pragma unroll
      for (int v = 0; v < 8; ++v){
        int m = m0 + i * 16 + v + 8 * (lane >> 4);
        float x = cc[v] + bv;
        if (act == 1) x = siluf(x);
        if (outF) outF[(size_t)m * N + n] = x;
        if (outB) outB[(size_t)m * N + n] = (bf16_t)x;
      }
    }
  }
}

// ---------------- WMMA GEMM, N == 64 ----------------
// block tile 64x64; 8 waves, wave tile 16x32 (1 m-frag x 2 n-frags). Branch-free.
__global__ void k_gemm64(const bf16_t* __restrict__ A, const bf16_t* __restrict__ Bp,
                         const float* __restrict__ bias,
                         float* __restrict__ outF, bf16_t* __restrict__ outB,
                         int Kpad, int act){
  const int N = 64, NB = 4;
  const int lane = threadIdx.x & 31;
  const int w = threadIdx.x >> 5;
  const int m0 = blockIdx.y * 64 + (w >> 1) * 16;
  const int n0 = (w & 1) * 32;
  const int KB = Kpad / 32;
  const v8f zero = {0.f,0.f,0.f,0.f,0.f,0.f,0.f,0.f};
  v8f c0 = zero, c1 = zero;
  for (int kb = 0; kb < KB; ++kb){
    v16bf a  = load_a(A, m0, Kpad, kb * 32, lane);
    v16bf b0 = load_bp(Bp, kb, (n0 >> 4),     NB, lane);
    v16bf b1 = load_bp(Bp, kb, (n0 >> 4) + 1, NB, lane);
    c0 = wmma_bf16(a, b0, c0);
    c1 = wmma_bf16(a, b1, c1);
  }
#pragma unroll
  for (int j = 0; j < 2; ++j){
    int n = n0 + j * 16 + (lane & 15);
    float bv = bias ? bias[n] : 0.0f;
    v8f cc = (j == 0) ? c0 : c1;
#pragma unroll
    for (int v = 0; v < 8; ++v){
      int m = m0 + v + 8 * (lane >> 4);
      float x = cc[v] + bv;
      if (act == 1) x = siluf(x);
      if (outF) outF[(size_t)m * N + n] = x;
      if (outB) outB[(size_t)m * N + n] = (bf16_t)x;
    }
  }
}

// ---------------- fused GRU layer (persistent over T) ----------------
// One workgroup owns 32 batch rows; h lives in LDS across all 100 steps.
// Per step: r/z/i_n/h_n fragments accumulated per-wave, gate math in registers.
template<bool REV, bool WSEQ, int DIN>
__global__ void k_gru(const bf16_t* __restrict__ xs,           // (T,B,DIN) bf16
                      const bf16_t* __restrict__ WihP,         // packed (DIN x 768)
                      const bf16_t* __restrict__ WhhP,         // packed (256 x 768)
                      const float* __restrict__ b_ih, const float* __restrict__ b_hh,
                      bf16_t* __restrict__ hseq,               // (T,B,512) or null
                      bf16_t* __restrict__ hlast,              // (B,512) or null
                      int colOff){
  __shared__ bf16_t hbuf[32 * HENC];
  __shared__ bf16_t xbuf[32 * DIN];
  __shared__ float  sbi[3 * HENC], sbh[3 * HENC];
  const int tid = threadIdx.x, lane = tid & 31, w = tid >> 5;
  const int m0 = blockIdx.x * 32;
  const int mf = (w >> 2) * 16;        // wave's m sub-tile (16 rows)
  const int nfb = (w & 3) * 4;         // wave's 4 n-fragments (64 gate cols)
  const int NB = 48;                   // 768/16
  for (int i = tid; i < 3 * HENC; i += 256){ sbi[i] = b_ih[i]; sbh[i] = b_hh[i]; }
  for (int i = tid; i < 32 * HENC; i += 256) hbuf[i] = (bf16_t)0.0f;
  __syncthreads();

  const v8f zero = {0.f,0.f,0.f,0.f,0.f,0.f,0.f,0.f};
  v8f cr[4], cz[4], cin[4], chn[4];

  for (int tt = 0; tt < TWIN; ++tt){
    const int t = REV ? (TWIN - 1 - tt) : tt;
    // stage x tile (rows m0..m0+31 contiguous in (T,B,DIN))
    const bf16_t* xsrc = xs + ((size_t)t * BATCH + m0) * DIN;
    for (int i = tid * 8; i < 32 * DIN; i += 256 * 8)
      *(v8bf*)(xbuf + i) = *(const v8bf*)(xsrc + i);
    if (tt + 1 < TWIN){
      const int tn = REV ? (t - 1) : (t + 1);
      __builtin_prefetch(xs + ((size_t)tn * BATCH + m0) * DIN + (size_t)tid * ((32 * DIN) / 256), 0, 3);
    }
    __syncthreads();

#pragma unroll
    for (int g = 0; g < 4; ++g){ cr[g] = zero; cz[g] = zero; cin[g] = zero; chn[g] = zero; }

    // phase 1: x @ W_ih^T  -> r, z, i_n
    for (int kb = 0; kb < DIN / 32; ++kb){
      v16bf a = load_a(xbuf, mf, DIN, kb * 32, lane);
#pragma unroll
      for (int g = 0; g < 4; ++g){
        const int nf = nfb + g;
        cr[g]  = wmma_bf16(a, load_bp(WihP, kb, nf,      NB, lane), cr[g]);
        cz[g]  = wmma_bf16(a, load_bp(WihP, kb, nf + 16, NB, lane), cz[g]);
        cin[g] = wmma_bf16(a, load_bp(WihP, kb, nf + 32, NB, lane), cin[g]);
      }
    }
    // phase 2: h @ W_hh^T  -> r, z, h_n
    for (int kb = 0; kb < HENC / 32; ++kb){
      v16bf a = load_a(hbuf, mf, HENC, kb * 32, lane);
#pragma unroll
      for (int g = 0; g < 4; ++g){
        const int nf = nfb + g;
        cr[g]  = wmma_bf16(a, load_bp(WhhP, kb, nf,      NB, lane), cr[g]);
        cz[g]  = wmma_bf16(a, load_bp(WhhP, kb, nf + 16, NB, lane), cz[g]);
        chn[g] = wmma_bf16(a, load_bp(WhhP, kb, nf + 32, NB, lane), chn[g]);
      }
    }
    __syncthreads();   // all hbuf reads done before gate writes

    // gates + state update, fully register-local per owned (m,n) element
#pragma unroll
    for (int g = 0; g < 4; ++g){
      const int n = (nfb + g) * 16 + (lane & 15);
      const float br = sbi[n] + sbh[n];
      const float bz = sbi[n + HENC] + sbh[n + HENC];
      const float bin = sbi[n + 2 * HENC];
      const float bhn = sbh[n + 2 * HENC];
#pragma unroll
      for (int v = 0; v < 8; ++v){
        const int mr = mf + v + 8 * (lane >> 4);
        const float r  = sigf(cr[g][v] + br);
        const float zg = sigf(cz[g][v] + bz);
        const float nn = tanhf(cin[g][v] + bin + r * (chn[g][v] + bhn));
        const float hold = (float)hbuf[mr * HENC + n];
        const float hn = (1.0f - zg) * nn + zg * hold;
        hbuf[mr * HENC + n] = (bf16_t)hn;
        if (WSEQ)
          hseq[((size_t)t * BATCH + m0 + mr) * (2 * HENC) + colOff + n] = (bf16_t)hn;
      }
    }
    __syncthreads();   // gate writes visible before next phase-2 reads
  }

  if (hlast){
    for (int i = tid; i < 32 * HENC; i += 256){
      const int rr = i >> 8, cc = i & (HENC - 1);
      hlast[(size_t)(m0 + rr) * (2 * HENC) + colOff + cc] = hbuf[i];
    }
  }
}

// ---------------- host ----------------

extern "C" void kernel_launch(void* const* d_in, const int* in_sizes, int n_in,
                              void* d_out, int out_size, void* d_ws, size_t ws_size,
                              hipStream_t stream){
  (void)in_sizes; (void)n_in; (void)ws_size;
  const float* x       = (const float*)d_in[0];
  // GRU: layer0 fwd(1..4) bwd(5..8), layer1 fwd(9..12) bwd(13..16)
  const float* Wih0f = (const float*)d_in[1],  *Whh0f = (const float*)d_in[2];
  const float* bih0f = (const float*)d_in[3],  *bhh0f = (const float*)d_in[4];
  const float* Wih0b = (const float*)d_in[5],  *Whh0b = (const float*)d_in[6];
  const float* bih0b = (const float*)d_in[7],  *bhh0b = (const float*)d_in[8];
  const float* Wih1f = (const float*)d_in[9],  *Whh1f = (const float*)d_in[10];
  const float* bih1f = (const float*)d_in[11], *bhh1f = (const float*)d_in[12];
  const float* Wih1b = (const float*)d_in[13], *Whh1b = (const float*)d_in[14];
  const float* bih1b = (const float*)d_in[15], *bhh1b = (const float*)d_in[16];
  const float* encW = (const float*)d_in[17], *encb = (const float*)d_in[18];
  const float* oW1 = (const float*)d_in[19], *ob1 = (const float*)d_in[20];
  const float* oW2 = (const float*)d_in[21], *ob2 = (const float*)d_in[22];
  const float* oW3 = (const float*)d_in[23], *ob3 = (const float*)d_in[24];
  const float* dW1 = (const float*)d_in[25], *db1 = (const float*)d_in[26];
  const float* dW2 = (const float*)d_in[27], *db2 = (const float*)d_in[28];
  const float* dW3 = (const float*)d_in[29], *db3 = (const float*)d_in[30];

  float* xhat = (float*)d_out;                                   // B*T*D
  float* z0out = (float*)d_out + (size_t)BATCH * TWIN * DFEAT;   // B*LAT
  float* keout = (float*)d_out + out_size - 1;                   // scalar

  // workspace carve
  size_t off = 0;
  auto alloc = [&](size_t bytes){
    void* p = (char*)d_ws + off;
    off = (off + bytes + 255) & ~(size_t)255;
    return p;
  };
  bf16_t* xbf   = (bf16_t*)alloc((size_t)TWIN * BATCH * DFEAT * 2);
  bf16_t* h1seq = (bf16_t*)alloc((size_t)TWIN * BATCH * 512 * 2);
  bf16_t* hcat  = (bf16_t*)alloc((size_t)BATCH * 512 * 2);
  bf16_t* pWih0f = (bf16_t*)alloc(64 * 768 * 2);
  bf16_t* pWih0b = (bf16_t*)alloc(64 * 768 * 2);
  bf16_t* pWhh0f = (bf16_t*)alloc(256 * 768 * 2);
  bf16_t* pWhh0b = (bf16_t*)alloc(256 * 768 * 2);
  bf16_t* pWih1f = (bf16_t*)alloc(512 * 768 * 2);
  bf16_t* pWih1b = (bf16_t*)alloc(512 * 768 * 2);
  bf16_t* pWhh1f = (bf16_t*)alloc(256 * 768 * 2);
  bf16_t* pWhh1b = (bf16_t*)alloc(256 * 768 * 2);
  bf16_t* pEnc   = (bf16_t*)alloc(512 * 64 * 2);
  bf16_t* pO1    = (bf16_t*)alloc(96 * 256 * 2);
  bf16_t* pO2    = (bf16_t*)alloc(256 * 256 * 2);
  bf16_t* pO3    = (bf16_t*)alloc(256 * 64 * 2);
  bf16_t* pD1    = (bf16_t*)alloc(64 * 512 * 2);
  bf16_t* pD2    = (bf16_t*)alloc(512 * 512 * 2);
  bf16_t* pD3    = (bf16_t*)alloc(512 * 6400 * 2);
  bf16_t* ztbf = (bf16_t*)alloc((size_t)BATCH * 96 * 2);
  bf16_t* oh1  = (bf16_t*)alloc((size_t)BATCH * 256 * 2);
  bf16_t* oh2  = (bf16_t*)alloc((size_t)BATCH * 256 * 2);
  float*  kb1  = (float*)alloc((size_t)BATCH * LAT * 4);
  float*  kb2  = (float*)alloc((size_t)BATCH * LAT * 4);
  float*  kb3  = (float*)alloc((size_t)BATCH * LAT * 4);
  float*  kb4  = (float*)alloc((size_t)BATCH * LAT * 4);
  float*  kfin = (float*)alloc((size_t)BATCH * LAT * 4);
  float*  zcur = (float*)alloc((size_t)BATCH * LAT * 4);
  float*  vsq  = (float*)alloc((size_t)10 * BATCH * 4);
  bf16_t* zdec = (bf16_t*)alloc((size_t)BATCH * LAT * 2);
  bf16_t* dh1  = (bf16_t*)alloc((size_t)BATCH * 512 * 2);
  bf16_t* dh2  = (bf16_t*)alloc((size_t)BATCH * 512 * 2);

  auto pack = [&](const float* W, bf16_t* Bp, int N, int K, int Kpad){
    size_t total = (size_t)(Kpad / 32) * (N / 16) * 512;
    k_pack_b<<<(unsigned)((total + 255) / 256), 256, 0, stream>>>(W, Bp, N, K, Kpad);
  };
  auto gemm = [&](const bf16_t* A, const bf16_t* Bp, const float* bias,
                  float* oF, bf16_t* oB, int N, int Kpad, int act){
    if (N % 128 == 0){
      dim3 g(N / 128, BATCH / 64);
      k_gemm128<<<g, 256, 0, stream>>>(A, Bp, bias, oF, oB, N, Kpad, act);
    } else {
      dim3 g(1, BATCH / 64);
      k_gemm64<<<g, 256, 0, stream>>>(A, Bp, bias, oF, oB, Kpad, act);
    }
  };

  // ---- stage 0: casts + weight packing ----
  {
    size_t total = (size_t)BATCH * TWIN * DFEAT;
    k_cast_x<<<(unsigned)((total + 255) / 256), 256, 0, stream>>>(x, xbf);
  }
  pack(Wih0f, pWih0f, 768, 64, 64);   pack(Wih0b, pWih0b, 768, 64, 64);
  pack(Whh0f, pWhh0f, 768, 256, 256); pack(Whh0b, pWhh0b, 768, 256, 256);
  pack(Wih1f, pWih1f, 768, 512, 512); pack(Wih1b, pWih1b, 768, 512, 512);
  pack(Whh1f, pWhh1f, 768, 256, 256); pack(Whh1b, pWhh1b, 768, 256, 256);
  pack(encW, pEnc, 64, 512, 512);
  pack(oW1, pO1, 256, 65, 96);  pack(oW2, pO2, 256, 256, 256); pack(oW3, pO3, 64, 256, 256);
  pack(dW1, pD1, 512, 64, 64);  pack(dW2, pD2, 512, 512, 512); pack(dW3, pD3, 6400, 512, 512);

  // ---- stage 1: bidirectional GRU encoder ----
  const unsigned NG = BATCH / 32;
  k_gru<false, true, 64><<<NG, 256, 0, stream>>>(xbf, pWih0f, pWhh0f, bih0f, bhh0f, h1seq, nullptr, 0);
  k_gru<true,  true, 64><<<NG, 256, 0, stream>>>(xbf, pWih0b, pWhh0b, bih0b, bhh0b, h1seq, nullptr, 256);
  k_gru<false, false, 512><<<NG, 256, 0, stream>>>(h1seq, pWih1f, pWhh1f, bih1f, bhh1f, nullptr, hcat, 0);
  k_gru<true,  false, 512><<<NG, 256, 0, stream>>>(h1seq, pWih1b, pWhh1b, bih1b, bhh1b, nullptr, hcat, 256);

  // ---- stage 2: latent projection ----
  gemm(hcat, pEnc, encb, z0out, nullptr, 64, 512, 0);
  hipMemcpyAsync(zcur, z0out, (size_t)BATCH * LAT * 4, hipMemcpyDeviceToDevice, stream);

  // ---- stage 3: RK4 Neural ODE (9 steps) + KE ----
  const float dt = 1.0f / 9.0f;
  auto feval = [&](const float* zin, const float* kin, float alpha, float tv, float* kout){
    k_make_zt<<<(BATCH * 96 + 255) / 256, 256, 0, stream>>>(zin, kin, alpha, tv, ztbf);
    gemm(ztbf, pO1, ob1, nullptr, oh1, 256, 96, 1);
    gemm(oh1,  pO2, ob2, nullptr, oh2, 256, 256, 1);
    gemm(oh2,  pO3, ob3, kout, nullptr, 64, 256, 0);
  };
  for (int i = 0; i < 9; ++i){
    const float t0 = (float)i * dt;
    feval(zcur, nullptr, 0.0f, t0, kb1);
    k_vsq<<<(BATCH + 255) / 256, 256, 0, stream>>>(kb1, vsq + (size_t)i * BATCH);
    feval(zcur, kb1, 0.5f * dt, t0 + 0.5f * dt, kb2);
    feval(zcur, kb2, 0.5f * dt, t0 + 0.5f * dt, kb3);
    feval(zcur, kb3, dt, t0 + dt, kb4);
    k_rk4<<<(BATCH * LAT + 255) / 256, 256, 0, stream>>>(zcur, kb1, kb2, kb3, kb4, dt);
  }
  feval(zcur, nullptr, 0.0f, 1.0f, kfin);
  k_vsq<<<(BATCH + 255) / 256, 256, 0, stream>>>(kfin, vsq + (size_t)9 * BATCH);
  k_ke<<<1, 256, 0, stream>>>(vsq, keout, dt);

  // ---- stage 4: decoder ----
  k_cast<<<(BATCH * LAT + 255) / 256, 256, 0, stream>>>(zcur, zdec, (size_t)BATCH * LAT);
  gemm(zdec, pD1, db1, nullptr, dh1, 512, 64, 1);
  gemm(dh1,  pD2, db2, nullptr, dh2, 512, 512, 1);
  gemm(dh2,  pD3, db3, xhat, nullptr, 6400, 512, 0);
}
